// TaskSpecificGIMANSystem_33492154974684
// MI455X (gfx1250) — compile-verified, matrix operand order checked by reading, not run on hardware
//
#include <hip/hip_runtime.h>
#include <hip/hip_bf16.h>

typedef __attribute__((ext_vector_type(16))) _Float16 v16h;
typedef __attribute__((ext_vector_type(8)))  _Float16 v8h;
typedef __attribute__((ext_vector_type(8)))  float    v8f;
typedef __attribute__((ext_vector_type(4)))  float    v4f;
typedef __attribute__((ext_vector_type(4)))  int      v4i;

#define NN   4096
#define SPD  256
#define GED  8
#define TED  64
#define EE   32
#define DD3  96
#define LNEPS 1e-5f

// ---------------- wave32 helpers ----------------
__device__ __forceinline__ float wsum32(float v) {
#pragma unroll
  for (int o = 16; o; o >>= 1) v += __shfl_xor(v, o, 32);
  return v;
}
__device__ __forceinline__ float lrelu(float x) { return x > 0.f ? x : 0.2f * x; }

// ---------------- Kernel 1: embeddings -> combined (4096 x 96) ----------------
__global__ __launch_bounds__(256) void embed_kernel(
    const float* __restrict__ spatial, const float* __restrict__ genomic,
    const float* __restrict__ temporal,
    const float* __restrict__ sp_w, const float* __restrict__ sp_b,
    const float* __restrict__ sp_lg, const float* __restrict__ sp_lb,
    const float* __restrict__ ge_w, const float* __restrict__ ge_b,
    const float* __restrict__ ge_lg, const float* __restrict__ ge_lb,
    const float* __restrict__ te_w, const float* __restrict__ te_b,
    const float* __restrict__ te_lg, const float* __restrict__ te_lb,
    float* __restrict__ combined) {
  const int lane = threadIdx.x & 31;
  const int row  = blockIdx.x * 8 + (threadIdx.x >> 5);

  float s = sp_b[lane];
  const float* srow = spatial + (size_t)row * SPD;
#pragma unroll 8
  for (int k = 0; k < SPD; ++k) s += srow[k] * sp_w[k * EE + lane];

  float g = ge_b[lane];
  const float* grow = genomic + (size_t)row * GED;
#pragma unroll
  for (int k = 0; k < GED; ++k) g += grow[k] * ge_w[k * EE + lane];

  float t = te_b[lane];
  const float* trow = temporal + (size_t)row * TED;
#pragma unroll 8
  for (int k = 0; k < TED; ++k) t += trow[k] * te_w[k * EE + lane];

  {
    float m = wsum32(s) * (1.f / 32.f);
    float d = s - m;
    float v = wsum32(d * d) * (1.f / 32.f);
    s = fmaxf(d * rsqrtf(v + LNEPS) * sp_lg[lane] + sp_lb[lane], 0.f);
  }
  {
    float m = wsum32(g) * (1.f / 32.f);
    float d = g - m;
    float v = wsum32(d * d) * (1.f / 32.f);
    g = fmaxf(d * rsqrtf(v + LNEPS) * ge_lg[lane] + ge_lb[lane], 0.f);
  }
  {
    float m = wsum32(t) * (1.f / 32.f);
    float d = t - m;
    float v = wsum32(d * d) * (1.f / 32.f);
    t = fmaxf(d * rsqrtf(v + LNEPS) * te_lg[lane] + te_lb[lane], 0.f);
  }
  float* crow = combined + (size_t)row * DD3;
  crow[lane] = s;
  crow[EE + lane] = g;
  crow[2 * EE + lane] = t;
}

// ------- Kernel 2: h = combined @ gat_w ; f1/f2 ; write hT(f16) and f2T -------
__global__ __launch_bounds__(256) void gat_h_kernel(
    const float* __restrict__ combined, const float* __restrict__ gat_w,
    const float* __restrict__ gat_a,
    float* __restrict__ f1, float* __restrict__ f2T, _Float16* __restrict__ hT) {
  const int lane = threadIdx.x & 31;
  const int row  = blockIdx.x * 8 + (threadIdx.x >> 5);
  const float* crow = combined + (size_t)row * DD3;
  float h0 = 0.f, h1 = 0.f;
#pragma unroll 4
  for (int k = 0; k < DD3; ++k) {
    float c = crow[k];
    h0 += c * gat_w[k * 64 + lane];
    h1 += c * gat_w[k * 64 + 32 + lane];
  }
  const float a1 = gat_a[lane], a2 = gat_a[32 + lane];
  float f10 = wsum32(h0 * a1);
  float f11 = wsum32(h1 * a1);
  float f20 = wsum32(h0 * a2);
  float f21 = wsum32(h1 * a2);
  if (lane == 0) {
    f1[row * 2 + 0] = f10;
    f1[row * 2 + 1] = f11;
    f2T[row]      = f20;
    f2T[NN + row] = f21;
  }
  // hT layout: [head][d][j]  (B-matrix friendly: contiguous in j)
  hT[(size_t)lane * NN + row]        = (_Float16)h0;
  hT[(size_t)(32 + lane) * NN + row] = (_Float16)h1;
}

// ---------------- Kernel 3: dense GAT attention via WMMA ----------------
__device__ __forceinline__ v16h loadB(const _Float16* __restrict__ p) {
  v8h lo = *(const v8h*)(p);        // K = kb .. kb+7
  v8h hi = *(const v8h*)(p + 16);   // K = kb+16 .. kb+23
  v16h b;
#pragma unroll
  for (int e = 0; e < 8; ++e) { b[e] = lo[e]; b[e + 8] = hi[e]; }
  return b;
}

__global__ __launch_bounds__(128) void gat_attn_kernel(
    const int* __restrict__ adj, const float* __restrict__ f1,
    const float* __restrict__ f2T, const _Float16* __restrict__ hT,
    float* __restrict__ h_out) {
  const int lane = threadIdx.x & 31;
  const int rb   = blockIdx.x * 4 + (threadIdx.x >> 5);  // 16-row block id
  const int i0   = rb * 16;
  const int M    = lane & 15;
  const int kb   = (lane < 16) ? 0 : 8;  // ISA 16-bit A-matrix lane K-base

  // global max of f2 per head: safe (>= masked max) softmax shift
  float mx0 = -1e30f, mx1 = -1e30f;
  for (int j = lane; j < NN; j += 32) {
    mx0 = fmaxf(mx0, f2T[j]);
    mx1 = fmaxf(mx1, f2T[NN + j]);
  }
#pragma unroll
  for (int o = 16; o; o >>= 1) {
    mx0 = fmaxf(mx0, __shfl_xor(mx0, o, 32));
    mx1 = fmaxf(mx1, __shfl_xor(mx1, o, 32));
  }
  const float f1v0 = f1[(i0 + M) * 2 + 0];
  const float f1v1 = f1[(i0 + M) * 2 + 1];
  const float c0m  = lrelu(f1v0 + mx0);
  const float c1m  = lrelu(f1v1 + mx1);

  v8f acc00 = {}, acc01 = {}, acc10 = {}, acc11 = {};
  float z0 = 0.f, z1 = 0.f;

  const int* __restrict__ adjrow = adj + (size_t)(i0 + M) * NN;
  const _Float16* __restrict__ bp = hT + (size_t)(lane & 15) * NN;

  for (int j0 = 0; j0 < NN; j0 += 32) {
    if (j0 + 32 < NN) __builtin_prefetch(adjrow + j0 + 32 + kb, 0, 1);

    v4i q0 = *(const v4i*)(adjrow + j0 + kb);
    v4i q1 = *(const v4i*)(adjrow + j0 + kb + 4);
    v4i q2 = *(const v4i*)(adjrow + j0 + kb + 16);
    v4i q3 = *(const v4i*)(adjrow + j0 + kb + 20);
    v4f p00 = *(const v4f*)(f2T + j0 + kb);
    v4f p01 = *(const v4f*)(f2T + j0 + kb + 4);
    v4f p02 = *(const v4f*)(f2T + j0 + kb + 16);
    v4f p03 = *(const v4f*)(f2T + j0 + kb + 20);
    v4f p10 = *(const v4f*)(f2T + NN + j0 + kb);
    v4f p11 = *(const v4f*)(f2T + NN + j0 + kb + 4);
    v4f p12 = *(const v4f*)(f2T + NN + j0 + kb + 16);
    v4f p13 = *(const v4f*)(f2T + NN + j0 + kb + 20);

    v16h A0, A1;  // alpha tiles, one per head
#pragma unroll
    for (int e = 0; e < 16; ++e) {
      int   av  = (e < 4) ? q0[e] : (e < 8) ? q1[e - 4] : (e < 12) ? q2[e - 8] : q3[e - 12];
      float fa  = (e < 4) ? p00[e] : (e < 8) ? p01[e - 4] : (e < 12) ? p02[e - 8] : p03[e - 12];
      float fb  = (e < 4) ? p10[e] : (e < 8) ? p11[e - 4] : (e < 12) ? p12[e - 8] : p13[e - 12];
      float al0 = av ? __expf(lrelu(f1v0 + fa) - c0m) : 0.f;
      float al1 = av ? __expf(lrelu(f1v1 + fb) - c1m) : 0.f;
      z0 += al0;
      z1 += al1;
      A0[e] = (_Float16)al0;
      A1[e] = (_Float16)al1;
    }

    v16h B00 = loadB(bp + j0 + kb);                        // head0, N tile 0
    v16h B01 = loadB(bp + (size_t)16 * NN + j0 + kb);      // head0, N tile 1
    v16h B10 = loadB(bp + (size_t)32 * NN + j0 + kb);      // head1, N tile 0
    v16h B11 = loadB(bp + (size_t)48 * NN + j0 + kb);      // head1, N tile 1

    acc00 = __builtin_amdgcn_wmma_f32_16x16x32_f16(false, A0, false, B00, (short)0, acc00, false, false);
    acc01 = __builtin_amdgcn_wmma_f32_16x16x32_f16(false, A0, false, B01, (short)0, acc01, false, false);
    acc10 = __builtin_amdgcn_wmma_f32_16x16x32_f16(false, A1, false, B10, (short)0, acc10, false, false);
    acc11 = __builtin_amdgcn_wmma_f32_16x16x32_f16(false, A1, false, B11, (short)0, acc11, false, false);
  }

  // Z per row (lane m and m+16 hold same M): fold halves, then normalize + store.
  float z0t = z0 + __shfl_xor(z0, 16, 32);
  float z1t = z1 + __shfl_xor(z1, 16, 32);
  const int base = (lane < 16) ? 0 : 8;  // C/D layout: lanes>=16 hold M=8..15
  const int n    = lane & 15;
#pragma unroll
  for (int r = 0; r < 8; ++r) {
    float zz0 = __shfl(z0t, base + r, 32);
    float zz1 = __shfl(z1t, base + r, 32);
    float iv0 = zz0 > 0.f ? 1.f / zz0 : 0.f;
    float iv1 = zz1 > 0.f ? 1.f / zz1 : 0.f;
    float* orow = h_out + (size_t)(i0 + base + r) * 64;
    orow[n]      = acc00[r] * iv0;
    orow[16 + n] = acc01[r] * iv0;
    orow[32 + n] = acc10[r] * iv1;
    orow[48 + n] = acc11[r] * iv1;
  }
}

// ---------------- Kernel 4: gat_out LN -> mha tail -> towers ----------------
struct HeadParams {
  const float *h_out, *combined;
  const float *gat_ow, *gat_ob, *gat_lg, *gat_lb;
  const float *mha_iw, *mha_ib, *mha_ow, *mha_ob;
  const float *m1_w, *m1_b, *m1_lg, *m1_lb, *m2_w, *m2_b, *m2_lg, *m2_lb, *m3_w, *m3_b;
  const float *c1_w, *c1_b, *c1_lg, *c1_lb, *c2_w, *c2_b, *c2_lg, *c2_lb, *c3_w, *c3_b;
  float* out;
};

__device__ __forceinline__ float tower96(
    int lane, const float fin[3],
    const float* __restrict__ w1, const float* __restrict__ b1,
    const float* __restrict__ lg1, const float* __restrict__ lb1,
    const float* __restrict__ w2, const float* __restrict__ b2,
    const float* __restrict__ lg2, const float* __restrict__ lb2,
    const float* __restrict__ w3, const float* __restrict__ b3) {
  float t1a = b1[lane], t1b = b1[32 + lane];
#pragma unroll
  for (int t = 0; t < 3; ++t) {
#pragma unroll
    for (int k2 = 0; k2 < 32; ++k2) {
      float fk = __shfl(fin[t], k2, 32);
      const float* wr = w1 + (size_t)(t * 32 + k2) * 64;
      t1a += fk * wr[lane];
      t1b += fk * wr[32 + lane];
    }
  }
  float m = wsum32(t1a + t1b) * (1.f / 64.f);
  float da = t1a - m, db = t1b - m;
  float v = wsum32(da * da + db * db) * (1.f / 64.f);
  float inv = rsqrtf(v + LNEPS);
  t1a = fmaxf(da * inv * lg1[lane] + lb1[lane], 0.f);
  t1b = fmaxf(db * inv * lg1[32 + lane] + lb1[32 + lane], 0.f);

  float t2v = b2[lane];
#pragma unroll
  for (int k2 = 0; k2 < 32; ++k2) {
    float ka = __shfl(t1a, k2, 32);
    float kb = __shfl(t1b, k2, 32);
    t2v += ka * w2[k2 * 32 + lane] + kb * w2[(32 + k2) * 32 + lane];
  }
  m = wsum32(t2v) * (1.f / 32.f);
  float d = t2v - m;
  v = wsum32(d * d) * (1.f / 32.f);
  t2v = fmaxf(d * rsqrtf(v + LNEPS) * lg2[lane] + lb2[lane], 0.f);
  return wsum32(t2v * w3[lane]) + b3[0];
}

__global__ __launch_bounds__(256) void head_kernel(HeadParams p) {
  const int lane = threadIdx.x & 31;
  const int row  = blockIdx.x * 8 + (threadIdx.x >> 5);

  // gat_out = LN(h_out @ gat_ow + gat_ob)
  float g = p.gat_ob[lane];
  const float* hrow = p.h_out + (size_t)row * 64;
#pragma unroll 4
  for (int k = 0; k < 64; ++k) g += hrow[k] * p.gat_ow[k * 32 + lane];
  {
    float m = wsum32(g) * (1.f / 32.f);
    float d = g - m;
    float v = wsum32(d * d) * (1.f / 32.f);
    g = d * rsqrtf(v + LNEPS) * p.gat_lg[lane] + p.gat_lb[lane];
  }

  // v = tile(gat_out,3) @ mha_iw[:,192:288] + mha_ib[192:]
  float vd0 = p.mha_ib[192 + lane];
  float vd1 = p.mha_ib[192 + 32 + lane];
  float vd2 = p.mha_ib[192 + 64 + lane];
#pragma unroll
  for (int k = 0; k < DD3; ++k) {
    float gk = __shfl(g, k & 31, 32);
    const float* wr = p.mha_iw + (size_t)k * 288 + 192;
    vd0 += gk * wr[lane];
    vd1 += gk * wr[32 + lane];
    vd2 += gk * wr[64 + lane];
  }

  // attn = v @ mha_ow + mha_ob
  float at0 = p.mha_ob[lane], at1 = p.mha_ob[32 + lane], at2 = p.mha_ob[64 + lane];
  float vparts[3] = {vd0, vd1, vd2};
#pragma unroll
  for (int t = 0; t < 3; ++t) {
#pragma unroll
    for (int k2 = 0; k2 < 32; ++k2) {
      float vk = __shfl(vparts[t], k2, 32);
      const float* wr = p.mha_ow + (size_t)(t * 32 + k2) * 96;
      at0 += vk * wr[lane];
      at1 += vk * wr[32 + lane];
      at2 += vk * wr[64 + lane];
    }
  }

  const float* crow = p.combined + (size_t)row * DD3;
  float fin[3] = {at0 + crow[lane], at1 + crow[32 + lane], at2 + crow[64 + lane]};

  float motor = tower96(lane, fin, p.m1_w, p.m1_b, p.m1_lg, p.m1_lb,
                        p.m2_w, p.m2_b, p.m2_lg, p.m2_lb, p.m3_w, p.m3_b);
  float cog = tower96(lane, fin, p.c1_w, p.c1_b, p.c1_lg, p.c1_lb,
                      p.c2_w, p.c2_b, p.c2_lg, p.c2_lb, p.c3_w, p.c3_b);
  cog = 1.f / (1.f + __expf(-cog));

  if (lane == 0) {
    p.out[row]      = motor;
    p.out[NN + row] = cog;
  }
}

// ---------------- host launcher ----------------
extern "C" void kernel_launch(void* const* d_in, const int* in_sizes, int n_in,
                              void* d_out, int out_size, void* d_ws, size_t ws_size,
                              hipStream_t stream) {
  const float* spatial  = (const float*)d_in[0];
  const float* genomic  = (const float*)d_in[1];
  const float* temporal = (const float*)d_in[2];
  const int*   adj      = (const int*)d_in[3];
  const float* sp_w = (const float*)d_in[4];
  const float* sp_b = (const float*)d_in[5];
  const float* sp_lg = (const float*)d_in[6];
  const float* sp_lb = (const float*)d_in[7];
  const float* ge_w = (const float*)d_in[8];
  const float* ge_b = (const float*)d_in[9];
  const float* ge_lg = (const float*)d_in[10];
  const float* ge_lb = (const float*)d_in[11];
  const float* te_w = (const float*)d_in[12];
  const float* te_b = (const float*)d_in[13];
  const float* te_lg = (const float*)d_in[14];
  const float* te_lb = (const float*)d_in[15];
  const float* gat_w = (const float*)d_in[16];
  const float* gat_a = (const float*)d_in[17];
  const float* gat_ow = (const float*)d_in[18];
  const float* gat_ob = (const float*)d_in[19];
  const float* gat_lg = (const float*)d_in[20];
  const float* gat_lb = (const float*)d_in[21];
  const float* mha_iw = (const float*)d_in[22];
  const float* mha_ib = (const float*)d_in[23];
  const float* mha_ow = (const float*)d_in[24];
  const float* mha_ob = (const float*)d_in[25];

  unsigned char* ws = (unsigned char*)d_ws;
  float*    combined = (float*)(ws);                 // 4096*96*4  = 1572864
  float*    f1       = (float*)(ws + 1572864);       // 4096*2*4   = 32768
  float*    f2T      = (float*)(ws + 1605632);       // 2*4096*4   = 32768
  _Float16* hT       = (_Float16*)(ws + 1638400);    // 2*32*4096*2= 524288
  float*    h_out    = (float*)(ws + 2162688);       // 4096*64*4  = 1048576

  embed_kernel<<<dim3(512), dim3(256), 0, stream>>>(
      spatial, genomic, temporal,
      sp_w, sp_b, sp_lg, sp_lb, ge_w, ge_b, ge_lg, ge_lb,
      te_w, te_b, te_lg, te_lb, combined);

  gat_h_kernel<<<dim3(512), dim3(256), 0, stream>>>(combined, gat_w, gat_a, f1, f2T, hT);

  gat_attn_kernel<<<dim3(64), dim3(128), 0, stream>>>(adj, f1, f2T, hT, h_out);

  HeadParams p;
  p.h_out = h_out;
  p.combined = combined;
  p.gat_ow = gat_ow; p.gat_ob = gat_ob; p.gat_lg = gat_lg; p.gat_lb = gat_lb;
  p.mha_iw = mha_iw; p.mha_ib = mha_ib; p.mha_ow = mha_ow; p.mha_ob = mha_ob;
  p.m1_w = (const float*)d_in[26]; p.m1_b = (const float*)d_in[27];
  p.m1_lg = (const float*)d_in[28]; p.m1_lb = (const float*)d_in[29];
  p.m2_w = (const float*)d_in[30]; p.m2_b = (const float*)d_in[31];
  p.m2_lg = (const float*)d_in[32]; p.m2_lb = (const float*)d_in[33];
  p.m3_w = (const float*)d_in[34]; p.m3_b = (const float*)d_in[35];
  p.c1_w = (const float*)d_in[36]; p.c1_b = (const float*)d_in[37];
  p.c1_lg = (const float*)d_in[38]; p.c1_lb = (const float*)d_in[39];
  p.c2_w = (const float*)d_in[40]; p.c2_b = (const float*)d_in[41];
  p.c2_lg = (const float*)d_in[42]; p.c2_lb = (const float*)d_in[43];
  p.c3_w = (const float*)d_in[44]; p.c3_b = (const float*)d_in[45];
  p.out = (float*)d_out;

  head_kernel<<<dim3(512), dim3(256), 0, stream>>>(p);
}
